// DecoderMultiHeadAttention_38293928411157
// MI455X (gfx1250) — matrix operand order repaired and checked by reading, not compiled, not run on hardware
//
#include <hip/hip_runtime.h>

// ---------------------------------------------------------------------------
// CDNA5 (gfx1250) decoder MHA, bf16 WMMA flash-attention pipeline.
//   D_MODEL=1024, H=16, DH=32, B=4, S=2048, scale = D_MODEL^-0.5 = 1/32
//   (scale is folded into Q at projection time)
// ---------------------------------------------------------------------------

#define D_MODEL 1024
#define NH      16
#define DHD     32
#define BATCH   4
#define SEQ     2048
#define NCHUNK  (SEQ / 32)
#define PER_TENSOR ((size_t)BATCH * NH * SEQ * DHD)   // 4.19M elements

typedef __attribute__((ext_vector_type(16))) __bf16 v16bf;
typedef __attribute__((ext_vector_type(8)))  __bf16 v8bf;
typedef __attribute__((ext_vector_type(8)))  float  v8f;

#define CONCAT16(lo, hi) \
    __builtin_shufflevector(lo, hi, 0, 1, 2, 3, 4, 5, 6, 7, 8, 9, 10, 11, 12, 13, 14, 15)

// ---------------------------------------------------------------------------
// Kernel 1: per-head QKV projection.  [S x 1024] * [1024 x 32] -> [S x 32]
// One wave = one 16x16 C tile; block = 8 waves = 64 rows x full DH.
// blockIdx.z selects q/k/v.  Q is pre-scaled by 1/32.  V is stored TRANSPOSED
// [B,H,DH,S] so the attention kernel's P*V B-operand is contiguous per lane.
// ---------------------------------------------------------------------------
__global__ __launch_bounds__(256)
void qkv_proj_kernel(const float* __restrict__ q, const float* __restrict__ k,
                     const float* __restrict__ v,
                     const float* __restrict__ Wq, const float* __restrict__ Wk,
                     const float* __restrict__ Wv,
                     __bf16* __restrict__ Qbf, __bf16* __restrict__ Kbf,
                     __bf16* __restrict__ Vt) {
    const int z    = blockIdx.z;
    const float* x = (z == 0) ? q : (z == 1) ? k : v;
    const float* W = (z == 0) ? Wq : (z == 1) ? Wk : Wv;

    const int lane = threadIdx.x & 31;
    const int wave = threadIdx.x >> 5;
    const int half = lane >> 4;     // A-layout: lanes 16-31 take K-offset +16
    const int l16  = lane & 15;

    const int bh = blockIdx.y;                       // b*16 + h
    const int h  = bh & (NH - 1);
    const int b  = bh >> 4;
    const int s0 = blockIdx.x * 64 + (wave >> 1) * 16;
    const int e0 = (wave & 1) * 16;

    const float* xrow  = x + (size_t)(b * SEQ + s0 + l16) * D_MODEL;
    const float* wbase = W + (size_t)h * D_MODEL * DHD;

    const int akoff = half * 8;
    v8f c = {};
    for (int kb = 0; kb < D_MODEL; kb += 32) {
        // Prefetch next K-step of the streaming A operand (global_prefetch_b8)
        if (kb + 32 < D_MODEL) __builtin_prefetch(xrow + kb + 32 + akoff, 0, 3);
        // A (activations): elems 0..15 -> K=16*half..16*half+15 (contiguous).
        // Two 32B loads + native bf16 cvt.
        typedef __attribute__((ext_vector_type(8))) float v8fv;
        v8fv alo = *(const v8fv*)(xrow + kb + 2 * akoff);
        v8fv ahi = *(const v8fv*)(xrow + kb + 2 * akoff + 8);
        v16bf av;
#pragma unroll
        for (int i = 0; i < 8; ++i) {
            av[i]     = (__bf16)alo[i];
            av[8 + i] = (__bf16)ahi[i];
        }
        // B (weights), B-layout: lane n holds col n; elem i = K row kb0+i.
        const int kb0 = kb + half * 16;
        v16bf bv;
#pragma unroll
        for (int i = 0; i < 16; ++i)
            bv[i] = (__bf16)wbase[(size_t)(kb0 + i) * DHD + e0 + l16];
        c = __builtin_amdgcn_wmma_f32_16x16x32_bf16(false, av, false, bv,
                                                    (short)0, c, false, false);
    }

    // C-layout: VGPR r -> row r + 8*half, col l16
    if (z == 2) {
        __bf16* o = Vt + (size_t)(bh * DHD + e0 + l16) * SEQ;   // [e][s]
#pragma unroll
        for (int r = 0; r < 8; ++r) o[s0 + r + 8 * half] = (__bf16)c[r];
    } else if (z == 0) {
        __bf16* o = Qbf + ((size_t)bh * SEQ + s0) * DHD;
#pragma unroll
        for (int r = 0; r < 8; ++r)   // fold softmax scale 1/sqrt(1024)
            o[(size_t)(r + 8 * half) * DHD + e0 + l16] = (__bf16)(c[r] * 0.03125f);
    } else {
        __bf16* o = Kbf + ((size_t)bh * SEQ + s0) * DHD;
#pragma unroll
        for (int r = 0; r < 8; ++r)
            o[(size_t)(r + 8 * half) * DHD + e0 + l16] = (__bf16)c[r];
    }
}

// ---------------------------------------------------------------------------
// Kernel 2: flash attention, double-buffered async staging.
// Each wave owns 16 query rows of one (b,h).  Per 32-key chunk the block
// cooperatively stages the K tile (32x32 bf16) and V tile (32x32 bf16) into
// LDS with GLOBAL_LOAD_ASYNC_TO_LDS_B128 (one 16B async transfer per thread).
// The stage for chunk i+1 is issued BEFORE computing chunk i, so the async
// DMA overlaps 2 WMMA (Q*K^T) + online softmax + 2 WMMA (P*V); only then
// s_wait_asynccnt + one barrier per iteration.  P does its C-layout ->
// A-layout transpose through a private per-wave LDS slab (same-wave DS ops
// are in-order, no barrier needed for that part).
// ---------------------------------------------------------------------------
__global__ __launch_bounds__(256)
void flash_attn_kernel(const __bf16* __restrict__ Qbf,
                       const __bf16* __restrict__ Kbf,
                       const __bf16* __restrict__ Vt,
                       __bf16* __restrict__ Xbf) {
    __shared__ __bf16 ktile[2][32 * 32];   // [buf][key][e]   2x2KB
    __shared__ __bf16 vtile[2][32 * 32];   // [buf][e][key]   2x2KB
    __shared__ __bf16 pbuf[8][16 * 32];    // per-wave P slabs, 8KB

    const int tid  = threadIdx.x;
    const int lane = tid & 31;
    const int wave = tid >> 5;
    const int half = lane >> 4;
    const int l16  = lane & 15;
    const int bh   = blockIdx.y;
    const int b    = bh >> 4;
    const int h    = bh & (NH - 1);
    const int s0   = blockIdx.x * 128 + wave * 16;

    // Q rows in A-layout (two contiguous 16B per-lane halves)
    const __bf16* qrow = Qbf + ((size_t)bh * SEQ + s0 + l16) * DHD;
    v16bf qa;
    {
        v8bf qlo = *(const v8bf*)(qrow + half * 16);
        v8bf qhi = *(const v8bf*)(qrow + half * 16 + 8);
        qa = CONCAT16(qlo, qhi);
    }

    float m[8], l[8];
    v8f o0 = {}, o1 = {};
#pragma unroll
    for (int r = 0; r < 8; ++r) { m[r] = -1e30f; l[r] = 0.0f; }

    // ---- async staging setup -------------------------------------------
    // Single block-uniform SGPR base: K slab for this (b,h).  The V slab
    // (transposed) lives exactly PER_TENSOR elements after the K slab in the
    // workspace, so V threads just bias their 32-bit voffset by 8MB.
    const __bf16* kgbase = Kbf + (size_t)bh * SEQ * DHD;      // [s][e]
    const unsigned long long sbase = (unsigned long long)(uintptr_t)kgbase;
    const bool doK  = (wave < 4);          // waves 0-3 stage K, 4-7 stage V
    const int  srow = (tid & 127) >> 2;    // 0..31 (key for K, e for V)
    const int  part = tid & 3;             // 16B chunk within 64B row
    const unsigned lds_dst0 = doK
        ? (unsigned)(uintptr_t)&ktile[0][srow * 32 + part * 8]
        : (unsigned)(uintptr_t)&vtile[0][srow * 32 + part * 8];
    const unsigned lds_dst1 = doK
        ? (unsigned)(uintptr_t)&ktile[1][srow * 32 + part * 8]
        : (unsigned)(uintptr_t)&vtile[1][srow * 32 + part * 8];
    const unsigned vbias = doK ? 0u : (unsigned)(PER_TENSOR * 2);   // bytes

    auto stage = [&](int kc, int buf) {
        unsigned voff = doK
            ? (unsigned)(((kc + srow) * DHD + part * 8) * 2)          // K bytes
            : vbias + (unsigned)((srow * SEQ + kc + part * 8) * 2);   // V bytes
        unsigned dst = buf ? lds_dst1 : lds_dst0;
        asm volatile("global_load_async_to_lds_b128 %0, %1, %2"
                     :: "v"(dst), "v"(voff), "s"(sbase) : "memory");
    };

    __bf16* pw = &pbuf[wave][0];

    // Prologue: stage chunk 0 into buffer 0
    stage(0, 0);
    asm volatile("s_wait_asynccnt 0x0" ::: "memory");
    __syncthreads();

    for (int i = 0; i < NCHUNK; ++i) {
        // Issue async stage for the NEXT chunk; overlaps with compute below.
        // (Previous iteration's barrier guarantees the alternate buffer is
        // no longer being read by any wave.)
        if (i + 1 < NCHUNK) stage((i + 1) * 32, (i + 1) & 1);

        const __bf16* kt = ktile[i & 1];
        const __bf16* vt = vtile[i & 1];

        // ---- Q*K^T: B-operand from LDS, lane n = key col ---------------
        v16bf kbt0 = *(const v16bf*)&kt[l16 * 32 + half * 16];
        v16bf kbt1 = *(const v16bf*)&kt[(16 + l16) * 32 + half * 16];
        v8f zc = {};
        v8f sv0 = __builtin_amdgcn_wmma_f32_16x16x32_bf16(false, qa, false, kbt0,
                                                          (short)0, zc, false, false);
        v8f sv1 = __builtin_amdgcn_wmma_f32_16x16x32_bf16(false, qa, false, kbt1,
                                                          (short)0, zc, false, false);

        // ---- online softmax: row r+8*half spans 16 lanes ---------------
#pragma unroll
        for (int r = 0; r < 8; ++r) {
            float x0 = sv0[r], x1 = sv1[r];
            float t = fmaxf(x0, x1);
            t = fmaxf(t, __shfl_xor(t, 1));
            t = fmaxf(t, __shfl_xor(t, 2));
            t = fmaxf(t, __shfl_xor(t, 4));
            t = fmaxf(t, __shfl_xor(t, 8));
            float mnew  = fmaxf(m[r], t);
            float alpha = __expf(m[r] - mnew);
            float p0 = __expf(x0 - mnew);
            float p1 = __expf(x1 - mnew);
            float rs = p0 + p1;
            rs += __shfl_xor(rs, 1);
            rs += __shfl_xor(rs, 2);
            rs += __shfl_xor(rs, 4);
            rs += __shfl_xor(rs, 8);
            l[r] = l[r] * alpha + rs;
            m[r] = mnew;
            o0[r] *= alpha;
            o1[r] *= alpha;
            const int row = r + 8 * half;
            pw[row * 32 + l16]      = (__bf16)p0;
            pw[row * 32 + 16 + l16] = (__bf16)p1;
        }

        // ---- reload P in A-layout (same wave: HW keeps DS in order) ----
        v16bf pa;
        {
            const __bf16* pr = pw + (size_t)l16 * 32 + half * 16;
            v8bf plo = *(const v8bf*)(pr);
            v8bf phi = *(const v8bf*)(pr + 8);
            pa = CONCAT16(plo, phi);
        }

        // ---- P*V: B-operand from LDS vtile [e][key] --------------------
        v16bf vb0 = *(const v16bf*)&vt[l16 * 32 + half * 16];
        v16bf vb1 = *(const v16bf*)&vt[(16 + l16) * 32 + half * 16];
        o0 = __builtin_amdgcn_wmma_f32_16x16x32_bf16(false, pa, false, vb0,
                                                     (short)0, o0, false, false);
        o1 = __builtin_amdgcn_wmma_f32_16x16x32_bf16(false, pa, false, vb1,
                                                     (short)0, o1, false, false);

        // Next chunk's staging (issued above) must be complete before any
        // wave starts reading the alternate buffer.
        asm volatile("s_wait_asynccnt 0x0" ::: "memory");
        __syncthreads();
    }

    // Normalize and write concat-head layout [b, s, h*32 + e] (bf16)
    __bf16* xout = Xbf + ((size_t)b * SEQ + s0) * (NH * DHD) + h * DHD;
#pragma unroll
    for (int r = 0; r < 8; ++r) {
        const int row = r + 8 * half;
        float inv = 1.0f / l[r];
        xout[(size_t)row * (NH * DHD) + l16]      = (__bf16)(o0[r] * inv);
        xout[(size_t)row * (NH * DHD) + 16 + l16] = (__bf16)(o1[r] * inv);
    }
}

// ---------------------------------------------------------------------------
// Kernel 3: output projection.  X [8192 x 512] bf16 * Wo [512 x 1024] -> fp32
// ---------------------------------------------------------------------------
__global__ __launch_bounds__(256)
void out_proj_kernel(const __bf16* __restrict__ Xbf,
                     const float* __restrict__ Wo, float* __restrict__ out) {
    const int lane = threadIdx.x & 31;
    const int wave = threadIdx.x >> 5;
    const int half = lane >> 4;
    const int l16  = lane & 15;
    const int m0 = blockIdx.x * 64 + (wave >> 1) * 16;
    const int n0 = blockIdx.y * 32 + (wave & 1) * 16;
    const int K  = NH * DHD;   // 512

    const __bf16* arow = Xbf + (size_t)(m0 + l16) * K;
    v8f c = {};
    for (int kb = 0; kb < K; kb += 32) {
        if (kb + 32 < K) __builtin_prefetch(arow + kb + 32, 0, 3);
        v8bf alo = *(const v8bf*)(arow + kb + half * 16);
        v8bf ahi = *(const v8bf*)(arow + kb + half * 16 + 8);
        v16bf av = CONCAT16(alo, ahi);
        const int kb0 = kb + half * 16;
        v16bf bv;
#pragma unroll
        for (int i = 0; i < 16; ++i)
            bv[i] = (__bf16)Wo[(size_t)(kb0 + i) * D_MODEL + n0 + l16];
        c = __builtin_amdgcn_wmma_f32_16x16x32_bf16(false, av, false, bv,
                                                    (short)0, c, false, false);
    }
    float* orow = out + (size_t)m0 * D_MODEL + n0 + l16;
#pragma unroll
    for (int r = 0; r < 8; ++r)
        orow[(size_t)(r + 8 * half) * D_MODEL] = c[r];
}

// ---------------------------------------------------------------------------
extern "C" void kernel_launch(void* const* d_in, const int* in_sizes, int n_in,
                              void* d_out, int out_size, void* d_ws, size_t ws_size,
                              hipStream_t stream) {
    const float* q  = (const float*)d_in[0];
    const float* k  = (const float*)d_in[1];
    const float* v  = (const float*)d_in[2];
    const float* Wq = (const float*)d_in[3];
    const float* Wk = (const float*)d_in[4];
    const float* Wv = (const float*)d_in[5];
    const float* Wo = (const float*)d_in[6];
    float* out = (float*)d_out;

    // Workspace layout (bf16): Q | K | Vt | X  — K and Vt MUST stay adjacent
    // (flash kernel addresses Vt as K-base + PER_TENSOR via async voffset).
    __bf16* Qbf = (__bf16*)d_ws;
    __bf16* Kbf = Qbf + PER_TENSOR;
    __bf16* Vt  = Kbf + PER_TENSOR;
    __bf16* Xbf = Vt + PER_TENSOR;   // [B, S, H*DH] concat layout

    dim3 g1(SEQ / 64, BATCH * NH, 3);
    qkv_proj_kernel<<<g1, 256, 0, stream>>>(q, k, v, Wq, Wk, Wv, Qbf, Kbf, Vt);

    dim3 g2(SEQ / 128, BATCH * NH, 1);
    flash_attn_kernel<<<g2, 256, 0, stream>>>(Qbf, Kbf, Vt, Xbf);

    dim3 g3((BATCH * SEQ) / 64, D_MODEL / 32, 1);
    out_proj_kernel<<<g3, 256, 0, stream>>>(Xbf, Wo, out);
}
